// STGAT_77704548319625
// MI455X (gfx1250) — matrix operand order repaired and checked by reading, not compiled, not run on hardware
//
#include <hip/hip_runtime.h>
#include <hip/hip_bf16.h>

typedef __attribute__((ext_vector_type(16))) _Float16 v16h;
typedef __attribute__((ext_vector_type(8)))  float    v8f;

union AF16 { v16h v; _Float16 e[16]; };
union CF8  { v8f  v; float    e[8];  };

__device__ __forceinline__ int akmap(int lane, int h) {
    // A-matrix (16-bit, 16x32) k index for element h: contiguous k = 16*(lane/16) + h (matches B fills)
    return ((lane >> 4) << 4) | h;
}
__device__ __forceinline__ v8f wmma_f16(v16h a, v16h b, v8f c) {
    return __builtin_amdgcn_wmma_f32_16x16x32_f16(false, a, false, b, (short)0, c, false, false);
}
// Branchless fast activations: raw v_exp_f32 / v_rcp_f32, no EXEC divergence.
__device__ __forceinline__ float fsig(float x) {
    return __builtin_amdgcn_rcpf(1.0f + __builtin_amdgcn_exp2f(x * -1.44269504f));
}
__device__ __forceinline__ float ftanh(float x) {
    // tanh(x) = 1 - 2/(1 + e^{2x})
    return 1.0f - 2.0f * __builtin_amdgcn_rcpf(1.0f + __builtin_amdgcn_exp2f(x * 2.88539008f));
}

// ---------------------------------------------------------------------------
// K1: bidirectional LSTM. block = 16 batch rows x 1 direction, 128 threads.
// ---------------------------------------------------------------------------
__global__ __launch_bounds__(128) void k_lstm(
    const float* __restrict__ past,
    const float* __restrict__ w_ih_f, const float* __restrict__ w_hh_f, const float* __restrict__ b_f,
    const float* __restrict__ w_ih_b, const float* __restrict__ w_hh_b, const float* __restrict__ b_b,
    _Float16* __restrict__ hcat)
{
    const int dir = blockIdx.y;
    const float* w_ih = dir ? w_ih_b : w_ih_f;
    const float* w_hh = dir ? w_hh_b : w_hh_f;
    const float* bias = dir ? b_b   : b_f;

    __shared__ _Float16 hl[16 * 64];
    __shared__ float    xl[16 * 8 * 2];

    const int tid = threadIdx.x;
    const int wv = tid >> 5, lane = tid & 31;
    const int hi = lane >> 4, nl = lane & 15;
    const int j = wv * 16 + nl;   // hidden index owned by this lane

    {
        const float* src = past + (size_t)blockIdx.x * 256;
        for (int q = tid; q < 256; q += 128) xl[q] = src[q];
        for (int q = tid; q < 1024; q += 128) hl[q] = (_Float16)0.f;
    }

    // Resident w_hh B-fragments: 4 gate blocks x 2 k-steps
    v16h Bh[4][2];
    float wi0[4], wi1[4], bb[4];
    #pragma unroll
    for (int g = 0; g < 4; ++g) {
        int n = g * 64 + j;
        wi0[g] = w_ih[n * 2 + 0];
        wi1[g] = w_ih[n * 2 + 1];
        bb[g]  = bias[n];
        #pragma unroll
        for (int ks = 0; ks < 2; ++ks) {
            AF16 u;
            #pragma unroll
            for (int h = 0; h < 16; ++h) {
                int k = ks * 32 + hi * 16 + h;
                u.e[h] = (_Float16)w_hh[n * 64 + k];
            }
            Bh[g][ks] = u.v;
        }
    }

    float cst[8], hv[8];
    #pragma unroll
    for (int r = 0; r < 8; ++r) { cst[r] = 0.f; hv[r] = 0.f; }

    for (int t = 0; t < 8; ++t) {
        int tt = dir ? (7 - t) : t;
        __syncthreads();                 // previous h writes visible
        AF16 a0, a1;
        #pragma unroll
        for (int h = 0; h < 16; ++h) {
            int k0 = akmap(lane, h);
            a0.e[h] = hl[nl * 64 + k0];
            a1.e[h] = hl[nl * 64 + 32 + k0];
        }
        __syncthreads();                 // all reads done before rewrites
        float gate[4][8];
        #pragma unroll
        for (int g = 0; g < 4; ++g) {
            CF8 acc;
            #pragma unroll
            for (int r = 0; r < 8; ++r) acc.e[r] = 0.f;
            acc.v = wmma_f16(a0.v, Bh[g][0], acc.v);
            acc.v = wmma_f16(a1.v, Bh[g][1], acc.v);
            #pragma unroll
            for (int r = 0; r < 8; ++r) {
                int m = r + hi * 8;
                gate[g][r] = acc.e[r] + xl[(m * 8 + tt) * 2 + 0] * wi0[g]
                                      + xl[(m * 8 + tt) * 2 + 1] * wi1[g] + bb[g];
            }
        }
        #pragma unroll
        for (int r = 0; r < 8; ++r) {
            float ig = fsig(gate[0][r]);
            float fg = fsig(gate[1][r]);
            float gg = ftanh(gate[2][r]);
            float og = fsig(gate[3][r]);
            cst[r] = fg * cst[r] + ig * gg;
            hv[r]  = og * ftanh(cst[r]);
            hl[(r + hi * 8) * 64 + j] = (_Float16)hv[r];
        }
    }
    #pragma unroll
    for (int r = 0; r < 8; ++r) {
        int row = blockIdx.x * 16 + r + hi * 8;
        hcat[(size_t)row * 128 + dir * 64 + j] = (_Float16)hv[r];
    }
}

// ---------------------------------------------------------------------------
// K2: h_enc = hcat @ proj_W.T + proj_b ; Wh = h_enc @ gat_W.T. 64 rows/block.
// hcat tile staged into LDS with async global->LDS DMA (ASYNCcnt path).
// ---------------------------------------------------------------------------
__global__ __launch_bounds__(128) void k_proj_gat(
    const _Float16* __restrict__ hcat,
    const float* __restrict__ proj_W, const float* __restrict__ proj_b,
    const float* __restrict__ gat_W,
    _Float16* __restrict__ Wh)
{
    __shared__ _Float16 hc[64 * 128];
    __shared__ _Float16 he[64 * 64];
    const int tid = threadIdx.x, wv = tid >> 5, lane = tid & 31;
    const int hi = lane >> 4, nl = lane & 15;

    {
        // Pure f16 copy: 16 KB global -> LDS via GLOBAL_LOAD_ASYNC_TO_LDS_B128
        // (GVS mode: saddr = base SGPR pair, vaddr = byte offset, vdst = LDS addr)
        const _Float16* gsrc = hcat + (size_t)blockIdx.x * 64 * 128;
        unsigned ldsbase = (unsigned)(unsigned long long)(void*)hc;  // low 32 bits = LDS byte addr
        #pragma unroll
        for (int it = 0; it < 8; ++it) {
            int q = tid + it * 128;                // 1024 x 16B chunks
            unsigned lds_addr = ldsbase + (unsigned)q * 16u;
            unsigned goff = (unsigned)q * 16u;
            asm volatile("global_load_async_to_lds_b128 %0, %1, %2 offset:0"
                         :: "v"(lds_addr), "v"(goff), "s"(gsrc)
                         : "memory");
        }
        asm volatile("s_wait_asynccnt 0" ::: "memory");
    }
    __syncthreads();

    const int mloc = wv * 16 + nl;
    AF16 a[4];
    #pragma unroll
    for (int ks = 0; ks < 4; ++ks) {
        #pragma unroll
        for (int h = 0; h < 16; ++h) a[ks].e[h] = hc[mloc * 128 + ks * 32 + akmap(lane, h)];
    }
    for (int nt = 0; nt < 4; ++nt) {
        int n = nt * 16 + nl;
        CF8 acc;
        #pragma unroll
        for (int r = 0; r < 8; ++r) acc.e[r] = proj_b[n];
        #pragma unroll
        for (int ks = 0; ks < 4; ++ks) {
            AF16 b;
            #pragma unroll
            for (int h = 0; h < 16; ++h) b.e[h] = (_Float16)proj_W[n * 128 + ks * 32 + hi * 16 + h];
            acc.v = wmma_f16(a[ks].v, b.v, acc.v);
        }
        #pragma unroll
        for (int r = 0; r < 8; ++r) he[(wv * 16 + r + hi * 8) * 64 + n] = (_Float16)acc.e[r];
    }
    __syncthreads();
    AF16 a2[2];
    #pragma unroll
    for (int ks = 0; ks < 2; ++ks) {
        #pragma unroll
        for (int h = 0; h < 16; ++h) a2[ks].e[h] = he[mloc * 64 + ks * 32 + akmap(lane, h)];
    }
    for (int nt = 0; nt < 4; ++nt) {
        int n = nt * 16 + nl;
        CF8 acc;
        #pragma unroll
        for (int r = 0; r < 8; ++r) acc.e[r] = 0.f;
        #pragma unroll
        for (int ks = 0; ks < 2; ++ks) {
            AF16 b;
            #pragma unroll
            for (int h = 0; h < 16; ++h) b.e[h] = (_Float16)gat_W[n * 64 + ks * 32 + hi * 16 + h];
            acc.v = wmma_f16(a2[ks].v, b.v, acc.v);
        }
        #pragma unroll
        for (int r = 0; r < 8; ++r) {
            int row = blockIdx.x * 64 + wv * 16 + r + hi * 8;
            Wh[(size_t)row * 64 + n] = (_Float16)acc.e[r];
        }
    }
}

// ---------------------------------------------------------------------------
// K3: p_j = exp(Wh_j . attn_w[64:128])  (row term + bias cancel in softmax)
// ---------------------------------------------------------------------------
__global__ __launch_bounds__(256) void k_score(
    const _Float16* __restrict__ Wh, const float* __restrict__ attn_w,
    float* __restrict__ p)
{
    int j = blockIdx.x * 256 + threadIdx.x;
    float s = 0.f;
    #pragma unroll 8
    for (int n = 0; n < 64; ++n) s += (float)Wh[(size_t)j * 64 + n] * attn_w[64 + n];
    p[j] = __expf(s);
}

// ---------------------------------------------------------------------------
// K4: gfin = (sum_j p_j * Wh_j) / (sum_j p_j)   (deterministic fixed order)
// ---------------------------------------------------------------------------
__global__ __launch_bounds__(256) void k_gsum(
    const _Float16* __restrict__ Wh, const float* __restrict__ p,
    float* __restrict__ gfin)
{
    __shared__ float red[256];
    __shared__ float Ssh;
    int tid = threadIdx.x;
    float acc = 0.f;
    for (int j = tid; j < 4096; j += 256) acc += p[j];
    red[tid] = acc; __syncthreads();
    for (int s = 128; s > 0; s >>= 1) { if (tid < s) red[tid] += red[tid + s]; __syncthreads(); }
    if (tid == 0) Ssh = red[0];
    __syncthreads();
    float S = Ssh;
    int n = tid & 63, part = tid >> 6;
    acc = 0.f;
    for (int j = part; j < 4096; j += 4) acc += p[j] * (float)Wh[(size_t)j * 64 + n];
    __syncthreads();
    red[tid] = acc; __syncthreads();
    if (tid < 64) gfin[tid] = (red[tid] + red[tid + 64] + red[tid + 128] + red[tid + 192]) / S;
}

// ---------------------------------------------------------------------------
// K5: fused CNN: conv1(s2)+relu -> conv2(s2)+relu -> global mean. 1 img/block.
// ---------------------------------------------------------------------------
__global__ __launch_bounds__(128) void k_cnn(
    const float* __restrict__ maps,
    const float* __restrict__ c1w, const float* __restrict__ c1b,
    const float* __restrict__ c2w, const float* __restrict__ c2b,
    float* __restrict__ mapfeat)
{
    __shared__ _Float16 in0[3 * 64 * 64];     // 24 KB
    __shared__ _Float16 out1[16 * 32 * 32];   // 32 KB
    __shared__ float    red[128 * 2];
    const int tid = threadIdx.x, wv = tid >> 5, lane = tid & 31;
    const int hi = lane >> 4, nl = lane & 15;
    const int img = blockIdx.x;

    {
        // Vectorized f32 staging: 24 x b128 loads per thread, pack to f16 LDS.
        const float4* src4 = (const float4*)(maps + (size_t)img * 12288);
        for (int q = tid; q < 3072; q += 128) {
            if (q + 128 < 3072) __builtin_prefetch(src4 + q + 128, 0, 0);
            float4 v = src4[q];
            in0[q * 4 + 0] = (_Float16)v.x;
            in0[q * 4 + 1] = (_Float16)v.y;
            in0[q * 4 + 2] = (_Float16)v.z;
            in0[q * 4 + 3] = (_Float16)v.w;
        }
    }
    // conv1 weights: k enumerates ic*9+ky*3+kx == row-major tail of OIHW
    AF16 b1;
    #pragma unroll
    for (int h = 0; h < 16; ++h) {
        int k = hi * 16 + h;
        b1.e[h] = (k < 27) ? (_Float16)c1w[nl * 27 + k] : (_Float16)0.f;
    }
    float bias1 = c1b[nl];
    __syncthreads();

    for (int tile = wv; tile < 64; tile += 4) {
        int pp = tile * 16 + nl, oy = pp >> 5, ox = pp & 31;
        AF16 a;
        #pragma unroll
        for (int h = 0; h < 16; ++h) {
            int k = akmap(lane, h);
            _Float16 v = (_Float16)0.f;
            if (k < 27) {
                int ic = k / 9, r9 = k - ic * 9, ky = r9 / 3, kx = r9 - ky * 3;
                int iy = oy * 2 + ky - 1, ix = ox * 2 + kx - 1;
                if ((unsigned)iy < 64u && (unsigned)ix < 64u) v = in0[(ic * 64 + iy) * 64 + ix];
            }
            a.e[h] = v;
        }
        CF8 acc;
        #pragma unroll
        for (int r = 0; r < 8; ++r) acc.e[r] = bias1;
        acc.v = wmma_f16(a.v, b1.v, acc.v);
        #pragma unroll
        for (int r = 0; r < 8; ++r) {
            float vv = fmaxf(acc.e[r], 0.f);
            out1[nl * 1024 + tile * 16 + r + hi * 8] = (_Float16)vv;
        }
    }
    __syncthreads();

    // conv2 weights: 2 oc-tiles x 5 k-steps (K=144 padded to 160)
    v16h B2[2][5];
    #pragma unroll
    for (int nt = 0; nt < 2; ++nt) {
        int oc = nt * 16 + nl;
        #pragma unroll
        for (int ks = 0; ks < 5; ++ks) {
            AF16 u;
            #pragma unroll
            for (int h = 0; h < 16; ++h) {
                int k = ks * 32 + hi * 16 + h;
                u.e[h] = (k < 144) ? (_Float16)c2w[oc * 144 + k] : (_Float16)0.f;
            }
            B2[nt][ks] = u.v;
        }
    }
    float ps0 = 0.f, ps1 = 0.f;
    float bias20 = c2b[nl], bias21 = c2b[16 + nl];
    for (int mt = wv; mt < 16; mt += 4) {
        int pp = mt * 16 + nl, oy = pp >> 4, ox = pp & 15;
        CF8 a0c, a1c;
        #pragma unroll
        for (int r = 0; r < 8; ++r) { a0c.e[r] = bias20; a1c.e[r] = bias21; }
        #pragma unroll
        for (int ks = 0; ks < 5; ++ks) {
            AF16 a;
            #pragma unroll
            for (int h = 0; h < 16; ++h) {
                int k = ks * 32 + akmap(lane, h);
                _Float16 v = (_Float16)0.f;
                if (k < 144) {
                    int ic = k / 9, r9 = k - ic * 9, ky = r9 / 3, kx = r9 - ky * 3;
                    int iy = oy * 2 + ky - 1, ix = ox * 2 + kx - 1;
                    if ((unsigned)iy < 32u && (unsigned)ix < 32u) v = out1[ic * 1024 + iy * 32 + ix];
                }
                a.e[h] = v;
            }
            a0c.v = wmma_f16(a.v, B2[0][ks], a0c.v);
            a1c.v = wmma_f16(a.v, B2[1][ks], a1c.v);
        }
        #pragma unroll
        for (int r = 0; r < 8; ++r) {
            ps0 += fmaxf(a0c.e[r], 0.f);
            ps1 += fmaxf(a1c.e[r], 0.f);
        }
    }
    red[tid * 2 + 0] = ps0; red[tid * 2 + 1] = ps1;
    __syncthreads();
    if (tid < 32) {
        int oc = tid, nn = oc & 15, g2 = oc >> 4;
        float s = 0.f;
        #pragma unroll
        for (int w2 = 0; w2 < 4; ++w2)
            #pragma unroll
            for (int gg = 0; gg < 2; ++gg)
                s += red[(w2 * 32 + gg * 16 + nn) * 2 + g2];
        mapfeat[(size_t)img * 32 + oc] = s * (1.f / 256.f);
    }
}

// ---------------------------------------------------------------------------
// K6: z = comb(gat const + map) ; out = dec(z) ; emit mu / exp(sigma) / pi
// ---------------------------------------------------------------------------
__global__ __launch_bounds__(128) void k_decode(
    const float* __restrict__ mapfeat, const float* __restrict__ gfin,
    const float* __restrict__ comb_W, const float* __restrict__ comb_b,
    const float* __restrict__ dec_W, const float* __restrict__ dec_b,
    float* __restrict__ out)
{
    __shared__ _Float16 zf[64 * 64];
    __shared__ float gl[64];
    __shared__ float pil[64 * 4];
    const int tid = threadIdx.x, wv = tid >> 5, lane = tid & 31;
    const int hi = lane >> 4, nl = lane & 15;
    const int rowBase = blockIdx.x * 64;
    if (tid < 64) gl[tid] = gfin[tid];
    __syncthreads();

    // stage 1: z[64 rows, 64] = const(GAT) + mapfeat @ comb_W[:,64:96].T
    AF16 a;
    #pragma unroll
    for (int h = 0; h < 16; ++h) {
        int k = akmap(lane, h);
        int row = rowBase + wv * 16 + nl;
        a.e[h] = (_Float16)mapfeat[(size_t)row * 32 + k];
    }
    for (int nt = 0; nt < 4; ++nt) {
        int n = nt * 16 + nl;
        float cv = comb_b[n];
        for (int k = 0; k < 64; ++k) cv += comb_W[n * 96 + k] * gl[k];
        AF16 b;
        #pragma unroll
        for (int h = 0; h < 16; ++h) b.e[h] = (_Float16)comb_W[n * 96 + 64 + hi * 16 + h];
        CF8 acc;
        #pragma unroll
        for (int r = 0; r < 8; ++r) acc.e[r] = cv;
        acc.v = wmma_f16(a.v, b.v, acc.v);
        #pragma unroll
        for (int r = 0; r < 8; ++r) zf[(wv * 16 + r + hi * 8) * 64 + n] = (_Float16)acc.e[r];
    }
    __syncthreads();

    AF16 a2[2];
    #pragma unroll
    for (int ks = 0; ks < 2; ++ks) {
        #pragma unroll
        for (int h = 0; h < 16; ++h) a2[ks].e[h] = zf[(wv * 16 + nl) * 64 + ks * 32 + akmap(lane, h)];
    }

    const size_t SGOFF = (size_t)4096 * 72, PIOFF = (size_t)2 * 4096 * 72;
    for (int nt = 0; nt < 10; ++nt) {
        int n = nt * 16 + nl;
        CF8 acc;
        float bb = (n < 147) ? dec_b[n] : 0.f;
        #pragma unroll
        for (int r = 0; r < 8; ++r) acc.e[r] = bb;
        #pragma unroll
        for (int ks = 0; ks < 2; ++ks) {
            AF16 b;
            #pragma unroll
            for (int h = 0; h < 16; ++h) {
                int k = ks * 32 + hi * 16 + h;
                b.e[h] = (n < 147) ? (_Float16)dec_W[n * 64 + k] : (_Float16)0.f;
            }
            acc.v = wmma_f16(a2[ks].v, b.v, acc.v);
        }
        #pragma unroll
        for (int r = 0; r < 8; ++r) {
            int lrow = wv * 16 + r + hi * 8;
            int i = rowBase + lrow;
            float v = acc.e[r];
            if (n < 72)       out[(size_t)i * 72 + n] = v;
            else if (n < 144) out[SGOFF + (size_t)i * 72 + (n - 72)] = __expf(v);
            else if (n < 147) pil[lrow * 4 + (n - 144)] = v;
        }
    }
    __syncthreads();
    if (tid < 64) {
        float e0 = pil[tid * 4], e1 = pil[tid * 4 + 1], e2 = pil[tid * 4 + 2];
        float mx = fmaxf(e0, fmaxf(e1, e2));
        float x0 = __expf(e0 - mx), x1 = __expf(e1 - mx), x2 = __expf(e2 - mx);
        float inv = 1.f / (x0 + x1 + x2);
        int i = rowBase + tid;
        out[PIOFF + (size_t)i * 3 + 0] = x0 * inv;
        out[PIOFF + (size_t)i * 3 + 1] = x1 * inv;
        out[PIOFF + (size_t)i * 3 + 2] = x2 * inv;
    }
}

extern "C" void kernel_launch(void* const* d_in, const int* in_sizes, int n_in,
                              void* d_out, int out_size, void* d_ws, size_t ws_size,
                              hipStream_t stream) {
    const float* past   = (const float*)d_in[0];
    const float* maps   = (const float*)d_in[1];
    const float* w_ih_f = (const float*)d_in[2];
    const float* w_hh_f = (const float*)d_in[3];
    const float* b_f    = (const float*)d_in[4];
    const float* w_ih_b = (const float*)d_in[5];
    const float* w_hh_b = (const float*)d_in[6];
    const float* b_b    = (const float*)d_in[7];
    const float* proj_W = (const float*)d_in[8];
    const float* proj_b = (const float*)d_in[9];
    const float* gat_W  = (const float*)d_in[10];
    const float* attn_w = (const float*)d_in[11];
    // d_in[12] (attn_b) cancels under the row softmax — unused
    const float* c1w    = (const float*)d_in[13];
    const float* c1b    = (const float*)d_in[14];
    const float* c2w    = (const float*)d_in[15];
    const float* c2b    = (const float*)d_in[16];
    const float* comb_W = (const float*)d_in[17];
    const float* comb_b = (const float*)d_in[18];
    const float* dec_W  = (const float*)d_in[19];
    const float* dec_b  = (const float*)d_in[20];

    char* ws = (char*)d_ws;
    _Float16* hcat    = (_Float16*)(ws + 0);          // 4096*128 f16 = 1 MB
    _Float16* Wh      = (_Float16*)(ws + 1048576);    // 4096*64 f16 = 512 KB
    float*    p       = (float*)(ws + 1572864);       // 4096 f32
    float*    gfin    = (float*)(ws + 1589248);       // 64 f32
    float*    mapfeat = (float*)(ws + 1589760);       // 4096*32 f32
    float*    out     = (float*)d_out;

    k_lstm    <<<dim3(256, 2), 128, 0, stream>>>(past, w_ih_f, w_hh_f, b_f, w_ih_b, w_hh_b, b_b, hcat);
    k_cnn     <<<4096, 128, 0, stream>>>(maps, c1w, c1b, c2w, c2b, mapfeat);
    k_proj_gat<<<64, 128, 0, stream>>>(hcat, proj_W, proj_b, gat_W, Wh);
    k_score   <<<16, 256, 0, stream>>>(Wh, attn_w, p);
    k_gsum    <<<1, 256, 0, stream>>>(Wh, p, gfin);
    k_decode  <<<64, 128, 0, stream>>>(mapfeat, gfin, comb_W, comb_b, dec_W, dec_b, out);
}
